// BridgeAttentionLayer_61718680043528
// MI455X (gfx1250) — matrix operand order repaired and checked by reading, not compiled, not run on hardware
//
#include <hip/hip_runtime.h>
#include <hip/hip_bf16.h>

typedef __attribute__((ext_vector_type(16))) _Float16 v16h;
typedef __attribute__((ext_vector_type(8)))  _Float16 v8h;
typedef __attribute__((ext_vector_type(8)))  float    v8f;
typedef __attribute__((ext_vector_type(4)))  int      v4i;

#define DIM 1024
#define HEADS 16
#define HD 64
#define NTOK 4096   // B*T = 4*1024
#define SEQ 1024

// ---------------------------------------------------------------------------
// CDNA5 async global->LDS copy (ASYNCcnt path), with sync fallback
// ---------------------------------------------------------------------------
#if defined(__HIP_DEVICE_COMPILE__) && defined(__has_builtin)
#  if __has_builtin(__builtin_amdgcn_global_load_async_to_lds_b128)
#    define USE_ASYNC_LDS 1
#  endif
#endif

#define AS1 __attribute__((address_space(1)))
#define AS3 __attribute__((address_space(3)))

__device__ __forceinline__ void copy16_to_lds(const _Float16* g, _Float16* l) {
#ifdef USE_ASYNC_LDS
  __builtin_amdgcn_global_load_async_to_lds_b128((AS1 v4i*)g, (AS3 v4i*)l, 0, 0);
#else
  *(v8h*)l = *(const v8h*)g;
#endif
}
__device__ __forceinline__ void async_wait() {
#ifdef USE_ASYNC_LDS
#  if __has_builtin(__builtin_amdgcn_s_wait_asynccnt)
  __builtin_amdgcn_s_wait_asynccnt(0);
#  else
  asm volatile("s_wait_asynccnt 0x0" ::: "memory");
#  endif
#endif
}

// ---------------------------------------------------------------------------
// WMMA helpers (gfx1250, wave32). A: 16x32 f16 (M in lanes, K in halves),
// B: 32x16 f16 (N in lanes, K in halves), C/D: 16x16 f32 (M in VGPRs, N lanes)
// ---------------------------------------------------------------------------
__device__ __forceinline__ v8f wmma16(v16h a, v16h b, v8f c) {
  return __builtin_amdgcn_wmma_f32_16x16x32_f16(false, a, false, b,
                                                (short)0, c, false, false);
}

// A fragment: lanes 0-15: m=lane, k in {0..7,16..23};
// lanes 16-31: m=lane-16, k in {8..15,24..31}.
__device__ __forceinline__ v16h load_a16x32(const _Float16* base, int ld) {
  int lane = threadIdx.x & 31;
  const _Float16* p = base + (size_t)(lane & 15) * ld + ((lane & 16) ? 8 : 0);
  union { v16h v; v8h h[2]; } u;
  u.h[0] = *(const v8h*)p;
  u.h[1] = *(const v8h*)(p + 16);
  return u.v;
}

// B fragment from K-major rows[n][k]: lanes 0-15: n=lane, k in {0..7,16..23};
// lanes 16-31: n=lane-16, k in {8..15,24..31}.
__device__ __forceinline__ v16h load_b32x16(const _Float16* base, int ld) {
  int lane = threadIdx.x & 31;
  const _Float16* p = base + (size_t)(lane & 15) * ld + ((lane & 16) ? 8 : 0);
  union { v16h v; v8h h[2]; } u;
  u.h[0] = *(const v8h*)p;
  u.h[1] = *(const v8h*)(p + 16);
  return u.v;
}

// ---------------------------------------------------------------------------
// LayerNorm (cols = 1024) -> f16
// ---------------------------------------------------------------------------
__global__ __launch_bounds__(256) void ln_cast_kernel(
    const float* __restrict__ x, const float* __restrict__ w,
    const float* __restrict__ b, _Float16* __restrict__ out) {
  __shared__ float rs[256], rs2[256];
  int row = blockIdx.x, tid = threadIdx.x;
  const float* xr = x + (size_t)row * DIM;
  float s = 0.f, s2 = 0.f;
#pragma unroll
  for (int i = 0; i < 4; i++) { float v = xr[tid + i * 256]; s += v; s2 += v * v; }
  rs[tid] = s; rs2[tid] = s2;
  __syncthreads();
  for (int st = 128; st > 0; st >>= 1) {
    if (tid < st) { rs[tid] += rs[tid + st]; rs2[tid] += rs2[tid + st]; }
    __syncthreads();
  }
  float mean = rs[0] * (1.f / DIM);
  float var  = rs2[0] * (1.f / DIM) - mean * mean;
  float inv  = rsqrtf(var + 1e-5f);
#pragma unroll
  for (int i = 0; i < 4; i++) {
    int c = tid + i * 256;
    out[(size_t)row * DIM + c] = (_Float16)((xr[c] - mean) * inv * w[c] + b[c]);
  }
}

// cast + transpose: W[K][N] f32 -> WT[N][K] f16
__global__ __launch_bounds__(256) void castT_kernel(
    const float* __restrict__ W, _Float16* __restrict__ WT, int K, int N) {
  size_t idx = (size_t)blockIdx.x * 256 + threadIdx.x;
  if (idx >= (size_t)K * N) return;
  size_t n = idx / K, k = idx % K;
  WT[idx] = (_Float16)W[k * N + n];
}

// ---------------------------------------------------------------------------
// RoPE in place on q/k regions of fused qkv buffer (row stride 3*DIM)
// ---------------------------------------------------------------------------
__global__ __launch_bounds__(256) void rope_kernel(_Float16* __restrict__ qkv) {
  size_t idx = (size_t)blockIdx.x * 256 + threadIdx.x;  // (tok, head, pair)
  int pair = idx & 31;
  int h    = (idx >> 5) & 15;
  int tok  = (int)(idx >> 9);
  if (tok >= NTOK) return;
  float pos = (float)(tok & (SEQ - 1));
  int d0 = 2 * pair, d1 = d0 + 1;
  const float lnb = 9.210340371976184f / 32.0f;  // ln(10000)/32
  float f0 = __expf(-(float)(d0 & 31) * lnb);
  float f1 = __expf(-(float)(d1 & 31) * lnb);
  float c0 = __cosf(pos * f0), s0 = __sinf(pos * f0);
  float c1 = __cosf(pos * f1), s1 = __sinf(pos * f1);
  _Float16* q = qkv + (size_t)tok * (3 * DIM) + h * HD;
  _Float16* k = q + DIM;
  float q0 = (float)q[d0], q1 = (float)q[d1];
  q[d0] = (_Float16)(q0 * c0 - q1 * s0);
  q[d1] = (_Float16)(q1 * c1 + q0 * s1);
  float k0 = (float)k[d0], k1 = (float)k[d1];
  k[d0] = (_Float16)(k0 * c0 - k1 * s0);
  k[d1] = (_Float16)(k1 * c1 + k0 * s1);
}

// ---------------------------------------------------------------------------
// WMMA GEMM: C[M][N] = A[M][K] (f16) * BT[N][K]^T (f16) + bias, epilogue:
//   epi 0: store f16 ; epi 1: +resid, store f32 ; epi 2: GELU, store f16
// block 256 = 8 waves, tile 128x128, BK = 32; wave tile 32x64 = 8 WMMAs.
// Double-buffered LDS, next tile staged via async global->LDS (ASYNCcnt)
// overlapping the current tile's WMMAs; one barrier per K-step.
// ---------------------------------------------------------------------------
__global__ __launch_bounds__(256) void gemm_wmma_kernel(
    const _Float16* __restrict__ A, const _Float16* __restrict__ BT,
    const float* __restrict__ bias, const float* __restrict__ resid,
    void* __restrict__ outp, int M, int N, int K, int epi) {
  __shared__ __align__(16) _Float16 As[2][128][40];  // 80B stride, conflict-free
  __shared__ __align__(16) _Float16 Bs[2][128][40];
  int tid = threadIdx.x, lane = tid & 31, w = tid >> 5;
  int m0 = blockIdx.y * 128, n0 = blockIdx.x * 128;
  int wm = (w & 3) * 32, wn = (w >> 2) * 64;
  v8f acc[2][4] = {};

  int r = tid >> 1, c = (tid & 1) * 16;
  const _Float16* gA = A + (size_t)(m0 + r) * K + c;
  const _Float16* gB = BT + (size_t)(n0 + r) * K + c;

  // prologue: stage K-tile 0 into buffer 0
  copy16_to_lds(gA,     &As[0][r][c]);
  copy16_to_lds(gA + 8, &As[0][r][c + 8]);
  copy16_to_lds(gB,     &Bs[0][r][c]);
  copy16_to_lds(gB + 8, &Bs[0][r][c + 8]);
  async_wait();
  __syncthreads();

  int cur = 0;
  for (int k0 = 0; k0 < K; k0 += 32) {
    if (k0 + 32 < K) {  // stage next K-tile into the other buffer (overlapped)
      const _Float16* nA = gA + k0 + 32;
      const _Float16* nB = gB + k0 + 32;
      copy16_to_lds(nA,     &As[cur ^ 1][r][c]);
      copy16_to_lds(nA + 8, &As[cur ^ 1][r][c + 8]);
      copy16_to_lds(nB,     &Bs[cur ^ 1][r][c]);
      copy16_to_lds(nB + 8, &Bs[cur ^ 1][r][c + 8]);
      __builtin_prefetch(gA + k0 + 64, 0, 1);
      __builtin_prefetch(gB + k0 + 64, 0, 1);
    }
    v16h af[2], bf[4];
#pragma unroll
    for (int i = 0; i < 2; i++) af[i] = load_a16x32(&As[cur][wm + i * 16][0], 40);
#pragma unroll
    for (int j = 0; j < 4; j++) bf[j] = load_b32x16(&Bs[cur][wn + j * 16][0], 40);
#pragma unroll
    for (int i = 0; i < 2; i++)
#pragma unroll
      for (int j = 0; j < 4; j++) acc[i][j] = wmma16(af[i], bf[j], acc[i][j]);
    async_wait();     // own async copies into next buffer complete
    __syncthreads();  // everyone done reading cur / writing next
    cur ^= 1;
  }

  int mloc = (lane & 16) >> 1;  // 0 or 8
#pragma unroll
  for (int i = 0; i < 2; i++) {
#pragma unroll
    for (int j = 0; j < 4; j++) {
      int ncol = n0 + wn + j * 16 + (lane & 15);
      float bv = bias[ncol];
#pragma unroll
      for (int rr = 0; rr < 8; rr++) {
        int mrow = m0 + wm + i * 16 + rr + mloc;
        float v = acc[i][j][rr] + bv;
        size_t idx = (size_t)mrow * N + ncol;
        if (epi == 1) {
          ((float*)outp)[idx] = v + resid[idx];
        } else if (epi == 2) {
          float g = 0.5f * v * (1.0f + erff(v * 0.70710678118654752f));
          ((_Float16*)outp)[idx] = (_Float16)g;
        } else {
          ((_Float16*)outp)[idx] = (_Float16)v;
        }
      }
    }
  }
}

// ---------------------------------------------------------------------------
// Flash attention over 2048 keys (first 1024 = self with q_self/rope,
// last 1024 = cross with q_cross). One wave = 16 query rows, block = 128 rows.
// ---------------------------------------------------------------------------
__global__ __launch_bounds__(256) void attn_kernel(
    const _Float16* __restrict__ qkv,  // [NTOK][3*DIM] (q|k|v, roped)
    const _Float16* __restrict__ qc,   // [NTOK][DIM]
    const _Float16* __restrict__ kv,   // [NTOK][2*DIM] (vggt_k|vggt_v)
    float* __restrict__ out) {         // [NTOK][DIM]
  __shared__ __align__(16) _Float16 VT[64][40];       // transposed V tile [d][key]
  __shared__ __align__(16) _Float16 Plds[8][16][40];  // per-wave P (C->A layout)
  int tid = threadIdx.x, lane = tid & 31, w = tid >> 5;
  int b = blockIdx.x >> 4, h = blockIdx.x & 15;
  size_t qtok0 = (size_t)b * SEQ + blockIdx.y * 128 + w * 16;

  v16h qs0 = load_a16x32(qkv + qtok0 * (3 * DIM) + h * HD, 3 * DIM);
  v16h qs1 = load_a16x32(qkv + qtok0 * (3 * DIM) + h * HD + 32, 3 * DIM);
  v16h qc0 = load_a16x32(qc + qtok0 * DIM + h * HD, DIM);
  v16h qc1 = load_a16x32(qc + qtok0 * DIM + h * HD + 32, DIM);

  float mrow[8], lrow[8];
  v8f acc[4] = {};
#pragma unroll
  for (int rr = 0; rr < 8; rr++) { mrow[rr] = -1e30f; lrow[rr] = 0.f; }

  for (int key0 = 0; key0 < 2 * SEQ; key0 += 32) {
    bool selfpart = key0 < SEQ;
    const _Float16 *Ksrc, *Vsrc;
    int ldk;
    v16h qa0, qa1;
    if (selfpart) {
      size_t kt = (size_t)b * SEQ + key0;
      Ksrc = qkv + kt * (3 * DIM) + DIM + h * HD;
      Vsrc = qkv + kt * (3 * DIM) + 2 * DIM + h * HD;
      ldk = 3 * DIM;
      qa0 = qs0; qa1 = qs1;
    } else {
      size_t kt = (size_t)b * SEQ + (key0 - SEQ);
      Ksrc = kv + kt * (2 * DIM) + h * HD;
      Vsrc = kv + kt * (2 * DIM) + DIM + h * HD;
      ldk = 2 * DIM;
      qa0 = qc0; qa1 = qc1;
    }
    __syncthreads();  // protect VT (WAR vs previous iteration)
    {  // cooperative transposed staging of 32x64 V tile
      int kk = tid >> 3, dc = (tid & 7) * 8;
      v8h vv = *(const v8h*)(Vsrc + (size_t)kk * ldk + dc);
#pragma unroll
      for (int i = 0; i < 8; i++) VT[dc + i][kk] = vv[i];
    }
    __syncthreads();

    // scores: two 16-key tiles, head_dim 64 as 2 WMMAs each
    v8f s[2];
#pragma unroll
    for (int t = 0; t < 2; t++) {
      v8f st = {};
      st = wmma16(qa0, load_b32x16(Ksrc + (size_t)(t * 16) * ldk, ldk), st);
      st = wmma16(qa1, load_b32x16(Ksrc + (size_t)(t * 16) * ldk + 32, ldk), st);
      s[t] = st;
    }

    // online softmax over this 32-key chunk
#pragma unroll
    for (int rr = 0; rr < 8; rr++) {
      float x0 = s[0][rr] * 0.125f, x1 = s[1][rr] * 0.125f;  // 1/sqrt(64)
      float mx = fmaxf(x0, x1);
#pragma unroll
      for (int off = 1; off < 16; off <<= 1) mx = fmaxf(mx, __shfl_xor(mx, off, 32));
      float mn = fmaxf(mrow[rr], mx);
      float corr = __expf(mrow[rr] - mn);
      float p0 = __expf(x0 - mn), p1 = __expf(x1 - mn);
      float rs = p0 + p1;
#pragma unroll
      for (int off = 1; off < 16; off <<= 1) rs += __shfl_xor(rs, off, 32);
      lrow[rr] = lrow[rr] * corr + rs;
      mrow[rr] = mn;
#pragma unroll
      for (int ch = 0; ch < 4; ch++) acc[ch][rr] *= corr;
      int m = rr + ((lane & 16) >> 1);
      Plds[w][m][lane & 15] = (_Float16)p0;
      Plds[w][m][16 + (lane & 15)] = (_Float16)p1;
    }

    // O += P(16x32) @ V(32x64): per-wave LDS, DS ops in order within wave
    v16h pa = load_a16x32(&Plds[w][0][0], 40);
#pragma unroll
    for (int ch = 0; ch < 4; ch++)
      acc[ch] = wmma16(pa, load_b32x16(&VT[ch * 16][0], 40), acc[ch]);
  }

  int mloc = (lane & 16) >> 1;
#pragma unroll
  for (int ch = 0; ch < 4; ch++) {
#pragma unroll
    for (int rr = 0; rr < 8; rr++) {
      size_t tok = qtok0 + rr + mloc;
      out[tok * DIM + h * HD + ch * 16 + (lane & 15)] = acc[ch][rr] / lrow[rr];
    }
  }
}

// ---------------------------------------------------------------------------
extern "C" void kernel_launch(void* const* d_in, const int* in_sizes, int n_in,
                              void* d_out, int out_size, void* d_ws, size_t ws_size,
                              hipStream_t stream) {
  (void)in_sizes; (void)n_in; (void)out_size; (void)ws_size;
  const float* x       = (const float*)d_in[0];
  const float* vggt    = (const float*)d_in[1];
  const float* ln_q_w  = (const float*)d_in[2];
  const float* ln_q_b  = (const float*)d_in[3];
  const float* ln_kv_w = (const float*)d_in[4];
  const float* ln_kv_b = (const float*)d_in[5];
  const float* ln_o_w  = (const float*)d_in[6];
  const float* ln_o_b  = (const float*)d_in[7];
  const float* W_qkv   = (const float*)d_in[8];
  const float* b_qkv   = (const float*)d_in[9];
  const float* W_cq    = (const float*)d_in[10];
  const float* b_cq    = (const float*)d_in[11];
  const float* W_kv    = (const float*)d_in[12];
  const float* b_kv    = (const float*)d_in[13];
  const float* W_out   = (const float*)d_in[14];
  const float* b_out   = (const float*)d_in[15];
  const float* ln_f_w  = (const float*)d_in[16];
  const float* ln_f_b  = (const float*)d_in[17];
  const float* W_f1    = (const float*)d_in[18];
  const float* b_f1    = (const float*)d_in[19];
  const float* W_f2    = (const float*)d_in[20];
  const float* b_f2    = (const float*)d_in[21];
  float* outp = (float*)d_out;

  char* ws = (char*)d_ws;
  size_t off = 0;
  auto alloc = [&](size_t bytes) {
    void* p = ws + off;
    off = (off + bytes + 255) & ~(size_t)255;
    return p;
  };
  _Float16* xn    = (_Float16*)alloc((size_t)NTOK * DIM * 2);
  _Float16* vn    = (_Float16*)alloc((size_t)NTOK * DIM * 2);
  _Float16* wqkvT = (_Float16*)alloc((size_t)3 * DIM * DIM * 2);
  _Float16* wcqT  = (_Float16*)alloc((size_t)DIM * DIM * 2);
  _Float16* wkvT  = (_Float16*)alloc((size_t)2 * DIM * DIM * 2);
  _Float16* woutT = (_Float16*)alloc((size_t)DIM * DIM * 2);
  _Float16* wf1T  = (_Float16*)alloc((size_t)4 * DIM * DIM * 2);
  _Float16* wf2T  = (_Float16*)alloc((size_t)4 * DIM * DIM * 2);
  _Float16* qkvb  = (_Float16*)alloc((size_t)NTOK * 3 * DIM * 2);
  _Float16* qcb   = (_Float16*)alloc((size_t)NTOK * DIM * 2);
  _Float16* kvb   = (_Float16*)alloc((size_t)NTOK * 2 * DIM * 2);
  float*    attn  = (float*)   alloc((size_t)NTOK * DIM * 4);
  _Float16* ln1   = (_Float16*)alloc((size_t)NTOK * DIM * 2);
  float*    x1    = (float*)   alloc((size_t)NTOK * DIM * 4);
  _Float16* ln2   = (_Float16*)alloc((size_t)NTOK * DIM * 2);
  _Float16* h1    = (_Float16*)alloc((size_t)NTOK * 4 * DIM * 2);

  auto castT = [&](const float* W, _Float16* WT, int K, int N) {
    size_t tot = (size_t)K * N;
    castT_kernel<<<(unsigned)((tot + 255) / 256), 256, 0, stream>>>(W, WT, K, N);
  };
  auto gemm = [&](const _Float16* A, const _Float16* BT, const float* bias,
                  const float* resid, void* out, int M, int N, int K, int epi) {
    dim3 grid(N / 128, M / 128);
    gemm_wmma_kernel<<<grid, 256, 0, stream>>>(A, BT, bias, resid, out, M, N, K, epi);
  };

  // weight prep (f32 -> f16, transposed to [N][K])
  castT(W_qkv, wqkvT, DIM, 3 * DIM);
  castT(W_cq,  wcqT,  DIM, DIM);
  castT(W_kv,  wkvT,  DIM, 2 * DIM);
  castT(W_out, woutT, DIM, DIM);
  castT(W_f1,  wf1T,  DIM, 4 * DIM);
  castT(W_f2,  wf2T,  4 * DIM, DIM);

  // input layernorms
  ln_cast_kernel<<<NTOK, 256, 0, stream>>>(x, ln_q_w, ln_q_b, xn);
  ln_cast_kernel<<<NTOK, 256, 0, stream>>>(vggt, ln_kv_w, ln_kv_b, vn);

  // projections
  gemm(xn, wqkvT, b_qkv, nullptr, qkvb, NTOK, 3 * DIM, DIM, 0);
  gemm(xn, wcqT,  b_cq,  nullptr, qcb,  NTOK, DIM,     DIM, 0);
  gemm(vn, wkvT,  b_kv,  nullptr, kvb,  NTOK, 2 * DIM, DIM, 0);

  // RoPE on q_self/k_self
  rope_kernel<<<(NTOK * HEADS * 32) / 256, 256, 0, stream>>>(qkvb);

  // flash attention over concat(self, cross) keys
  attn_kernel<<<dim3(4 * HEADS, SEQ / 128), 256, 0, stream>>>(qkvb, qcb, kvb, attn);

  // out projection + residual
  ln_cast_kernel<<<NTOK, 256, 0, stream>>>(attn, ln_o_w, ln_o_b, ln1);
  gemm(ln1, woutT, b_out, x, x1, NTOK, DIM, DIM, 1);

  // FFN + residual
  ln_cast_kernel<<<NTOK, 256, 0, stream>>>(x1, ln_f_w, ln_f_b, ln2);
  gemm(ln2, wf1T, b_f1, nullptr, h1, NTOK, 4 * DIM, DIM, 2);
  gemm(h1, wf2T, b_f2, x1, outp, NTOK, DIM, 4 * DIM, 1);
}
